// MoELayer_24111946400448
// MI455X (gfx1250) — matrix operand order repaired and checked by reading, not compiled, not run on hardware
//
#include <hip/hip_runtime.h>
#include <hip/hip_bf16.h>

// ---------------------------------------------------------------------------
// MoE layer for MI455X (gfx1250): dense 8-expert SwiGLU MoE, bf16 WMMA path.
// Compute-bound (AI ~1900 flop/byte). GEMMs use LDS-staged, double-buffered
// tiles filled with GLOBAL_LOAD_ASYNC_TO_LDS_B128 (ASYNCcnt) so global loads
// are coalesced 16B/lane chunks; WMMA fragments are gathered from LDS.
// ---------------------------------------------------------------------------

#define N_TOK  4096
#define DMODEL 1024
#define HFRAC  2048
#define HSW    4096
#define NEXP   8
#define NFRAC  4

#define BM 64     // block tile M
#define BN 128    // block tile N
#define BK 64     // block K step (2 WMMA k-steps per stage)

typedef __bf16 bf16;
typedef bf16  bf16x8  __attribute__((ext_vector_type(8)));
typedef bf16  bf16x16 __attribute__((ext_vector_type(16)));
typedef float v8f     __attribute__((ext_vector_type(8)));

// ---- bf16 <-> f32 via bit ops (round-to-nearest-even) ----------------------
__device__ __forceinline__ bf16 f2bf(float f) {
  unsigned u = __builtin_bit_cast(unsigned, f);
  u += 0x7FFFu + ((u >> 16) & 1u);
  unsigned short h = (unsigned short)(u >> 16);
  return __builtin_bit_cast(bf16, h);
}
__device__ __forceinline__ float bf2f(bf16 b) {
  unsigned short h = __builtin_bit_cast(unsigned short, b);
  unsigned u = ((unsigned)h) << 16;
  return __builtin_bit_cast(float, u);
}

// ---- async global -> LDS copy (16 bytes per lane) --------------------------
// Generic pointers into LDS have the LDS byte offset in addr[31:0] (ISA 10.2).
__device__ __forceinline__ unsigned lds_off(const void* p) {
  return (unsigned)(size_t)p;
}
__device__ __forceinline__ void async_copy16(unsigned lds_addr, const bf16* g) {
  asm volatile("global_load_async_to_lds_b128 %0, %1, off"
               :: "v"(lds_addr), "v"((unsigned long long)(size_t)g)
               : "memory");
}
__device__ __forceinline__ void wait_async0() {
  asm volatile("s_wait_asynccnt 0" ::: "memory");
}

// ---- WMMA fragment gathers from LDS ---------------------------------------
// A matrix 16x32 bf16: lanes 0-15 row=lane, K 0..15;
// lanes 16-31 row=lane-16, K 16..31.
__device__ __forceinline__ bf16x16 lds_frag_a(const bf16* base, int ld, int lane) {
  const bf16* p = base + (lane & 15) * ld + ((lane >> 4) << 4);
  bf16x16 r;
  *(bf16x8*)(&r)       = *(const bf16x8*)(p);
  *((bf16x8*)(&r) + 1) = *(const bf16x8*)(p + 8);
  return r;
}
// B matrix 32x16: lane = column (lane&15); lanes 0-15 K 0..15, lanes 16-31
// K 16..31 (2 values per VGPR). Column n of B = row n of W (X @ W^T).
__device__ __forceinline__ bf16x16 lds_frag_b(const bf16* base, int ld, int lane) {
  const bf16* p = base + (lane & 15) * ld + ((lane >> 4) << 4);
  bf16x16 r;
  *(bf16x8*)(&r)       = *(const bf16x8*)(p);
  *((bf16x8*)(&r) + 1) = *(const bf16x8*)(p + 8);
  return r;
}
__device__ __forceinline__ v8f wmma_bf16(bf16x16 a, bf16x16 b, v8f c) {
  return __builtin_amdgcn_wmma_f32_16x16x32_bf16(
      false, a, false, b, (short)0, c, false, false);
}

// ---- trivial utility kernels ----------------------------------------------
__global__ __launch_bounds__(256) void zero_kernel(float* __restrict__ p, size_t n) {
  size_t i = (size_t)blockIdx.x * blockDim.x + threadIdx.x;
  size_t stride = (size_t)gridDim.x * blockDim.x;
  for (; i < n; i += stride) p[i] = 0.f;
}
__global__ __launch_bounds__(256) void cvt_kernel(const float* __restrict__ s,
                                                  bf16* __restrict__ d, size_t n) {
  size_t i = (size_t)blockIdx.x * blockDim.x + threadIdx.x;
  size_t stride = (size_t)gridDim.x * blockDim.x;
  for (; i < n; i += stride) d[i] = f2bf(s[i]);
}

// ---- router: one wave per token -------------------------------------------
__global__ __launch_bounds__(256) void router_kernel(
    const float* __restrict__ X, const float* __restrict__ RW,
    float* __restrict__ combine) {
  const int lane  = threadIdx.x & 31;
  const int token = blockIdx.x * 8 + (threadIdx.x >> 5);
  float acc[NEXP];
#pragma unroll
  for (int e = 0; e < NEXP; ++e) acc[e] = 0.f;
  const float* xr = X + (size_t)token * DMODEL;
  for (int d = lane; d < DMODEL; d += 32) {
    float xv = xr[d];
#pragma unroll
    for (int e = 0; e < NEXP; ++e) acc[e] += xv * RW[(size_t)e * DMODEL + d];
  }
#pragma unroll
  for (int e = 0; e < NEXP; ++e)
#pragma unroll
    for (int off = 16; off > 0; off >>= 1)
      acc[e] += __shfl_xor(acc[e], off, 32);
  if (lane == 0) {
    float mx = acc[0];
#pragma unroll
    for (int e = 1; e < NEXP; ++e) mx = fmaxf(mx, acc[e]);
    float w[NEXP], s = 0.f;
#pragma unroll
    for (int e = 0; e < NEXP; ++e) { w[e] = __expf(acc[e] - mx); s += w[e]; }
#pragma unroll
    for (int e = 0; e < NEXP; ++e) w[e] /= s;
    int i1 = 0;
#pragma unroll
    for (int e = 1; e < NEXP; ++e) if (w[e] > w[i1]) i1 = e;
    int i2 = (i1 == 0) ? 1 : 0;
#pragma unroll
    for (int e = 0; e < NEXP; ++e) if (e != i1 && w[e] > w[i2]) i2 = e;
    float t = fmaxf(w[i1] + w[i2], 1e-9f);
#pragma unroll
    for (int e = 0; e < NEXP; ++e)
      combine[(size_t)token * NEXP + e] =
          (e == i1) ? w[i1] / t : (e == i2) ? w[i2] / t : 0.f;
  }
}

// ---- rmsnorm + bf16 conversion of activations ------------------------------
__global__ __launch_bounds__(256) void rmsnorm_kernel(
    const float* __restrict__ X, const float* __restrict__ FR,
    bf16* __restrict__ Xb, bf16* __restrict__ Xnb) {
  const int token = blockIdx.x;
  const int tid = threadIdx.x, lane = tid & 31, wid = tid >> 5;
  __shared__ float red[8];
  __shared__ float srinv;
  const float* xr = X + (size_t)token * DMODEL;
  float ss = 0.f;
  for (int d = tid; d < DMODEL; d += 256) {
    float v = xr[d];
    ss += v * v;
    Xb[(size_t)token * DMODEL + d] = f2bf(v);
  }
#pragma unroll
  for (int off = 16; off > 0; off >>= 1) ss += __shfl_xor(ss, off, 32);
  if (lane == 0) red[wid] = ss;
  __syncthreads();
  if (tid == 0) {
    float t = 0.f;
#pragma unroll
    for (int i = 0; i < 8; ++i) t += red[i];
    srinv = rsqrtf(t / (float)DMODEL + 1e-6f);
  }
  __syncthreads();
  float rinv = srinv;
  for (int e = 0; e < NFRAC; ++e)
    for (int d = tid; d < DMODEL; d += 256) {
      float v = xr[d] * rinv * FR[(size_t)e * DMODEL + d];
      Xnb[(size_t)e * N_TOK * DMODEL + (size_t)token * DMODEL + d] = f2bf(v);
    }
}

// ---- fused gate/up GEMM: H = silu(Xn @ W1^T) * (Xn @ W3^T) ------------------
// 8 waves tiled 2(M) x 4(N); wave tile 32x32; LDS double-buffered staging.
__global__ __launch_bounds__(256) void gemm13_kernel(
    const bf16* __restrict__ Xn, const bf16* __restrict__ W1,
    const bf16* __restrict__ W3, bf16* __restrict__ Hout,
    int K, int Hdim) {
  __shared__ bf16 sA [2][BM * BK];   //  8 KB x2
  __shared__ bf16 sB1[2][BN * BK];   // 16 KB x2
  __shared__ bf16 sB3[2][BN * BK];   // 16 KB x2  -> 80 KB total
  const int tid  = threadIdx.x;
  const int lane = tid & 31;
  const int wid  = tid >> 5;
  const int mblk = blockIdx.x * BM;
  const int nblk = blockIdx.y * BN;
  const int wm = (wid & 1) * 32;
  const int wn = (wid >> 1) * 32;

  auto issue_stage = [&](int kt, int buf) {
    const int k0 = kt * BK;
    // A tile: 64 rows x 64 cols = 512 chunks of 8 bf16 (16 B)
#pragma unroll
    for (int r = 0; r < 2; ++r) {
      int chunk = tid + r * 256;
      int row = chunk >> 3, cc = (chunk & 7) << 3;
      async_copy16(lds_off(&sA[buf][row * BK + cc]),
                   Xn + (size_t)(mblk + row) * K + k0 + cc);
    }
    // B tiles: 128 rows x 64 cols = 1024 chunks each
#pragma unroll
    for (int r = 0; r < 4; ++r) {
      int chunk = tid + r * 256;
      int row = chunk >> 3, cc = (chunk & 7) << 3;
      async_copy16(lds_off(&sB1[buf][row * BK + cc]),
                   W1 + (size_t)(nblk + row) * K + k0 + cc);
      async_copy16(lds_off(&sB3[buf][row * BK + cc]),
                   W3 + (size_t)(nblk + row) * K + k0 + cc);
    }
  };

  v8f g00 = {}, g01 = {}, g10 = {}, g11 = {};
  v8f u00 = {}, u01 = {}, u10 = {}, u11 = {};
  const int KT = K / BK;
  issue_stage(0, 0);
  for (int kt = 0; kt < KT; ++kt) {
    const int buf = kt & 1;
    wait_async0();         // stage kt resident in LDS
    __syncthreads();
    if (kt + 1 < KT) issue_stage(kt + 1, buf ^ 1);
    const bf16* A  = &sA [buf][0];
    const bf16* B1 = &sB1[buf][0];
    const bf16* B3 = &sB3[buf][0];
#pragma unroll
    for (int kk = 0; kk < BK; kk += 32) {
      bf16x16 a0  = lds_frag_a(A  + (wm +  0) * BK + kk, BK, lane);
      bf16x16 a1  = lds_frag_a(A  + (wm + 16) * BK + kk, BK, lane);
      bf16x16 b10 = lds_frag_b(B1 + (wn +  0) * BK + kk, BK, lane);
      bf16x16 b11 = lds_frag_b(B1 + (wn + 16) * BK + kk, BK, lane);
      bf16x16 b30 = lds_frag_b(B3 + (wn +  0) * BK + kk, BK, lane);
      bf16x16 b31 = lds_frag_b(B3 + (wn + 16) * BK + kk, BK, lane);
      g00 = wmma_bf16(a0, b10, g00);  g01 = wmma_bf16(a0, b11, g01);
      g10 = wmma_bf16(a1, b10, g10);  g11 = wmma_bf16(a1, b11, g11);
      u00 = wmma_bf16(a0, b30, u00);  u01 = wmma_bf16(a0, b31, u01);
      u10 = wmma_bf16(a1, b30, u10);  u11 = wmma_bf16(a1, b31, u11);
    }
    __syncthreads();       // all waves done reading before buffer reuse
  }

  const int m0 = mblk + wm, n0 = nblk + wn;
  auto store_tile = [&](v8f g, v8f u, int mi, int nj) {
    int n = n0 + nj * 16 + (lane & 15);
    int mbase = m0 + mi * 16 + ((lane >> 4) << 3);
#pragma unroll
    for (int v = 0; v < 8; ++v) {
      float gv = g[v], uv = u[v];
      float hv = gv / (1.f + __expf(-gv)) * uv;   // silu(g) * u
      Hout[(size_t)(mbase + v) * Hdim + n] = f2bf(hv);
    }
  };
  store_tile(g00, u00, 0, 0); store_tile(g01, u01, 0, 1);
  store_tile(g10, u10, 1, 0); store_tile(g11, u11, 1, 1);
}

// ---- down GEMM core (shared by fractal / switch epilogues) -----------------
#define GEMM2_PROLOG()                                                        \
  __shared__ bf16 sA[2][BM * BK];                                             \
  __shared__ bf16 sB[2][BN * BK];                                             \
  const int tid  = threadIdx.x;                                               \
  const int lane = tid & 31;                                                  \
  const int wid  = tid >> 5;                                                  \
  const int mblk = blockIdx.x * BM;                                           \
  const int nblk = blockIdx.y * BN;                                           \
  const int wm = (wid & 1) * 32;                                              \
  const int wn = (wid >> 1) * 32;                                             \
  auto issue_stage = [&](int kt, int buf) {                                   \
    const int k0 = kt * BK;                                                   \
    _Pragma("unroll")                                                         \
    for (int r = 0; r < 2; ++r) {                                             \
      int chunk = tid + r * 256;                                              \
      int row = chunk >> 3, cc = (chunk & 7) << 3;                            \
      async_copy16(lds_off(&sA[buf][row * BK + cc]),                          \
                   Hb + (size_t)(mblk + row) * K + k0 + cc);                  \
    }                                                                         \
    _Pragma("unroll")                                                         \
    for (int r = 0; r < 4; ++r) {                                             \
      int chunk = tid + r * 256;                                              \
      int row = chunk >> 3, cc = (chunk & 7) << 3;                            \
      async_copy16(lds_off(&sB[buf][row * BK + cc]),                          \
                   W2 + (size_t)(nblk + row) * K + k0 + cc);                  \
    }                                                                         \
  };                                                                          \
  v8f c00 = {}, c01 = {}, c10 = {}, c11 = {};                                 \
  const int KT = K / BK;                                                      \
  issue_stage(0, 0);                                                          \
  for (int kt = 0; kt < KT; ++kt) {                                           \
    const int buf = kt & 1;                                                   \
    wait_async0();                                                            \
    __syncthreads();                                                          \
    if (kt + 1 < KT) issue_stage(kt + 1, buf ^ 1);                            \
    const bf16* A = &sA[buf][0];                                              \
    const bf16* B = &sB[buf][0];                                              \
    _Pragma("unroll")                                                         \
    for (int kk = 0; kk < BK; kk += 32) {                                     \
      bf16x16 a0 = lds_frag_a(A + (wm +  0) * BK + kk, BK, lane);             \
      bf16x16 a1 = lds_frag_a(A + (wm + 16) * BK + kk, BK, lane);             \
      bf16x16 b0 = lds_frag_b(B + (wn +  0) * BK + kk, BK, lane);             \
      bf16x16 b1 = lds_frag_b(B + (wn + 16) * BK + kk, BK, lane);             \
      c00 = wmma_bf16(a0, b0, c00);  c01 = wmma_bf16(a0, b1, c01);            \
      c10 = wmma_bf16(a1, b0, c10);  c11 = wmma_bf16(a1, b1, c11);            \
    }                                                                         \
    __syncthreads();                                                          \
  }                                                                           \
  const int m0 = mblk + wm, n0 = nblk + wn;

// fractal: out += c * (gamma * (xn + h@W2^T) + x)
__global__ __launch_bounds__(256) void gemm2_frac_kernel(
    const bf16* __restrict__ Hb, const bf16* __restrict__ W2,
    const bf16* __restrict__ Xn, const float* __restrict__ X,
    const float* __restrict__ gamma, const float* __restrict__ combine,
    int expert, int K, float* __restrict__ Out) {
  GEMM2_PROLOG()
  auto emit = [&](v8f acc, int mi, int nj) {
    int n = n0 + nj * 16 + (lane & 15);
    int mbase = m0 + mi * 16 + ((lane >> 4) << 3);
    float gm = gamma[n];
#pragma unroll
    for (int v = 0; v < 8; ++v) {
      int m = mbase + v;
      size_t idx = (size_t)m * DMODEL + n;
      float c = combine[(size_t)m * NEXP + expert];
      Out[idx] += c * (gm * (bf2f(Xn[idx]) + acc[v]) + X[idx]);
    }
  };
  emit(c00, 0, 0); emit(c01, 0, 1); emit(c10, 1, 0); emit(c11, 1, 1);
}

// switch: out += c * (h @ W2^T)
__global__ __launch_bounds__(256) void gemm2_sw_kernel(
    const bf16* __restrict__ Hb, const bf16* __restrict__ W2,
    const float* __restrict__ combine, int expert, int K,
    float* __restrict__ Out) {
  GEMM2_PROLOG()
  auto emit = [&](v8f acc, int mi, int nj) {
    int n = n0 + nj * 16 + (lane & 15);
    int mbase = m0 + mi * 16 + ((lane >> 4) << 3);
#pragma unroll
    for (int v = 0; v < 8; ++v) {
      int m = mbase + v;
      size_t idx = (size_t)m * DMODEL + n;
      Out[idx] += combine[(size_t)m * NEXP + expert] * acc[v];
    }
  };
  emit(c00, 0, 0); emit(c01, 0, 1); emit(c10, 1, 0); emit(c11, 1, 1);
}

// ---------------------------------------------------------------------------
extern "C" void kernel_launch(void* const* d_in, const int* in_sizes, int n_in,
                              void* d_out, int out_size, void* d_ws, size_t ws_size,
                              hipStream_t stream) {
  (void)in_sizes; (void)n_in; (void)out_size; (void)ws_size;
  const float* x          = (const float*)d_in[0];
  const float* router_w   = (const float*)d_in[1];
  const float* frac_rms   = (const float*)d_in[2];
  const float* frac_w1    = (const float*)d_in[3];
  const float* frac_w2    = (const float*)d_in[4];
  const float* frac_w3    = (const float*)d_in[5];
  const float* frac_gamma = (const float*)d_in[6];
  const float* sw_w1      = (const float*)d_in[7];
  const float* sw_w2      = (const float*)d_in[8];
  const float* sw_w3      = (const float*)d_in[9];
  float* out = (float*)d_out;

  char* ws = (char*)d_ws;
  size_t off = 0;
  auto take = [&](size_t bytes) -> void* {
    void* p = ws + off;
    off = (off + bytes + 255) & ~(size_t)255;
    return p;
  };
  float* combine = (float*)take((size_t)N_TOK * NEXP * 4);
  bf16* xb   = (bf16*)take((size_t)N_TOK * DMODEL * 2);
  bf16* xnb  = (bf16*)take((size_t)NFRAC * N_TOK * DMODEL * 2);
  bf16* fw1b = (bf16*)take((size_t)NFRAC * HFRAC * DMODEL * 2);
  bf16* fw2b = (bf16*)take((size_t)NFRAC * DMODEL * HFRAC * 2);
  bf16* fw3b = (bf16*)take((size_t)NFRAC * HFRAC * DMODEL * 2);
  bf16* sw1b = (bf16*)take((size_t)NFRAC * HSW * DMODEL * 2);
  bf16* sw2b = (bf16*)take((size_t)NFRAC * DMODEL * HSW * 2);
  bf16* sw3b = (bf16*)take((size_t)NFRAC * HSW * DMODEL * 2);
  bf16* hbuf = (bf16*)take((size_t)N_TOK * HSW * 2);

  zero_kernel<<<512, 256, 0, stream>>>(out, (size_t)N_TOK * DMODEL);

  cvt_kernel<<<2048, 256, 0, stream>>>(frac_w1, fw1b, (size_t)NFRAC * HFRAC * DMODEL);
  cvt_kernel<<<2048, 256, 0, stream>>>(frac_w2, fw2b, (size_t)NFRAC * DMODEL * HFRAC);
  cvt_kernel<<<2048, 256, 0, stream>>>(frac_w3, fw3b, (size_t)NFRAC * HFRAC * DMODEL);
  cvt_kernel<<<4096, 256, 0, stream>>>(sw_w1, sw1b, (size_t)NFRAC * HSW * DMODEL);
  cvt_kernel<<<4096, 256, 0, stream>>>(sw_w2, sw2b, (size_t)NFRAC * DMODEL * HSW);
  cvt_kernel<<<4096, 256, 0, stream>>>(sw_w3, sw3b, (size_t)NFRAC * HSW * DMODEL);

  router_kernel<<<N_TOK / 8, 256, 0, stream>>>(x, router_w, combine);
  rmsnorm_kernel<<<N_TOK, 256, 0, stream>>>(x, frac_rms, xb, xnb);

  for (int e = 0; e < NFRAC; ++e) {
    gemm13_kernel<<<dim3(N_TOK / BM, HFRAC / BN), 256, 0, stream>>>(
        xnb + (size_t)e * N_TOK * DMODEL,
        fw1b + (size_t)e * HFRAC * DMODEL,
        fw3b + (size_t)e * HFRAC * DMODEL,
        hbuf, DMODEL, HFRAC);
    gemm2_frac_kernel<<<dim3(N_TOK / BM, DMODEL / BN), 256, 0, stream>>>(
        hbuf, fw2b + (size_t)e * DMODEL * HFRAC,
        xnb + (size_t)e * N_TOK * DMODEL, x,
        frac_gamma + (size_t)e * DMODEL, combine, e, HFRAC, out);
  }
  for (int j = 0; j < NFRAC; ++j) {
    gemm13_kernel<<<dim3(N_TOK / BM, HSW / BN), 256, 0, stream>>>(
        xb, sw1b + (size_t)j * HSW * DMODEL,
        sw3b + (size_t)j * HSW * DMODEL, hbuf, DMODEL, HSW);
    gemm2_sw_kernel<<<dim3(N_TOK / BM, DMODEL / BN), 256, 0, stream>>>(
        hbuf, sw2b + (size_t)j * DMODEL * HSW, combine, NFRAC + j, HSW, out);
  }
}
